// PositionFeaturizer_80796924772816
// MI455X (gfx1250) — matrix operand order repaired and checked by reading, not compiled, not run on hardware
//
#include <hip/hip_runtime.h>
#include <math.h>

typedef _Float16 half_t;
typedef __attribute__((ext_vector_type(16))) _Float16 v16h;
typedef __attribute__((ext_vector_type(8)))  _Float16 v8h;
typedef __attribute__((ext_vector_type(8)))  float    v8f;

constexpr int N_NODES = 16384;
constexpr int EMB     = 512;
constexpr int HID     = 1024;
constexpr int NHEAD   = 8;
constexpr int HDIM    = 64;
constexpr int DEG     = 16;
constexpr int KPAD    = 544;    // EMB + 3*H = 536, padded to multiple of 32
constexpr float SCALE = 0.125f; // 1/sqrt(64)

// ---------------------------------------------------------------------------
// fp32 -> f16 weight conversion + TRANSPOSE (+ zero K-padding).
// src: [srcK, cols] row-major fp32.  dst: [cols, dstK] row-major f16 (B^T).
// One-time per call; matrices are tiny (<= 2.2 MB) and L2-resident.
// ---------------------------------------------------------------------------
__global__ void cvt_wT_kernel(const float* __restrict__ src, half_t* __restrict__ dst,
                              int srcK, int dstK, int cols) {
    int idx = blockIdx.x * 256 + threadIdx.x;
    int total = cols * dstK;
    if (idx >= total) return;
    int c = idx / dstK;        // output row  = weight column
    int r = idx - c * dstK;    // output col  = weight row (K index)
    dst[idx] = (r < srcK) ? (half_t)src[(size_t)r * cols + c] : (half_t)0.0f;
}

// ---------------------------------------------------------------------------
// Dual LayerNorm: one block (128 threads) per row of x [N, 512].
// Writes z_att (f16, [N,512]) and h_buf[:, 0:512] (f16, ld=544); zeros pad cols.
// ---------------------------------------------------------------------------
__global__ __launch_bounds__(128)
void ln_dual_kernel(const float* __restrict__ x,
                    const float* __restrict__ ga, const float* __restrict__ ba,
                    const float* __restrict__ gm, const float* __restrict__ bm,
                    half_t* __restrict__ z_att, half_t* __restrict__ h_buf) {
    const int n = blockIdx.x;
    const int t = threadIdx.x;
    __shared__ float red[128];
    const float* row = x + (size_t)n * EMB;

    float v[4];
    float s = 0.f;
#pragma unroll
    for (int i = 0; i < 4; ++i) { v[i] = row[t + 128 * i]; s += v[i]; }
    red[t] = s; __syncthreads();
    for (int off = 64; off > 0; off >>= 1) {
        if (t < off) red[t] += red[t + off];
        __syncthreads();
    }
    const float mean = red[0] * (1.0f / EMB);
    __syncthreads();

    s = 0.f;
#pragma unroll
    for (int i = 0; i < 4; ++i) { float d = v[i] - mean; s += d * d; }
    red[t] = s; __syncthreads();
    for (int off = 64; off > 0; off >>= 1) {
        if (t < off) red[t] += red[t + off];
        __syncthreads();
    }
    const float rinv = rsqrtf(red[0] * (1.0f / EMB) + 1e-5f);

#pragma unroll
    for (int i = 0; i < 4; ++i) {
        int c = t + 128 * i;
        float zn = (v[i] - mean) * rinv;
        z_att[(size_t)n * EMB + c]  = (half_t)(zn * ga[c] + ba[c]);
        h_buf[(size_t)n * KPAD + c] = (half_t)(zn * gm[c] + bm[c]);
    }
    if (t < KPAD - (EMB + 3 * NHEAD))   // zero pad columns 536..543
        h_buf[(size_t)n * KPAD + EMB + 3 * NHEAD + t] = (half_t)0.0f;
}

// ---------------------------------------------------------------------------
// WMMA f16 GEMM: C[M,Ncols] = A[M,K] x B[K,Ncols], B given pre-transposed
// (Bt[Ncols,K], leading dim == K).
//   MODE 0: outH = f16(acc + bias)
//   MODE 1: outH = f16(gelu_tanh(acc + bias))
//   MODE 2: outF = acc + bias + resid
// Block tile 128x128, BK=32, 256 threads = 8 waves (2x4), wave tile 64x32
// -> 8 v_wmma_f32_16x16x32_f16 per wave per K-step.
// ---------------------------------------------------------------------------
constexpr int BM = 128, BN = 128, BK = 32;
constexpr int LDA_S = 48;  // halves: 96B rows -> 32B-aligned 16/32B chunks
constexpr int LDB_S = 48;

__device__ __forceinline__ float gelu_tanh(float v) {
    return 0.5f * v * (1.0f + tanhf(0.7978845608028654f * (v + 0.044715f * v * v * v)));
}

template <int MODE>
__global__ __launch_bounds__(256)
void gemm_f16_kernel(const half_t* __restrict__ A, int lda,
                     const half_t* __restrict__ Bt,          // [Ncols, K]
                     const float* __restrict__ bias,
                     const float* __restrict__ resid,
                     half_t* __restrict__ outH,
                     float* __restrict__ outF,
                     int M, int Ncols, int K) {
    __shared__ half_t As[BM * LDA_S];   // 12 KB
    __shared__ half_t Bs[BN * LDB_S];   // 12 KB (rows = output cols, contiguous K)

    const int t    = threadIdx.x;
    const int wave = t >> 5;
    const int lane = t & 31;
    const int wm   = wave >> 2;        // 0..1  (64-row slab)
    const int wn   = wave & 3;         // 0..3  (32-col slab)
    const int bm0  = blockIdx.x * BM;
    const int bn0  = blockIdx.y * BN;

    const int lrow  = lane & 15;
    const int lhalf = lane >> 4;

    v8f acc[4][2];
    const v8f zv = {0.f, 0.f, 0.f, 0.f, 0.f, 0.f, 0.f, 0.f};
#pragma unroll
    for (int i = 0; i < 4; ++i)
#pragma unroll
        for (int j = 0; j < 2; ++j) acc[i][j] = zv;

    // staging: 256 threads, each moves one 32B chunk of A and one of Bt
    const int sRow   = t >> 1;          // 0..127
    const int sChunk = (t & 1) * 16;    // 0 / 16 halves

    const half_t* aSrc = A  + (size_t)(bm0 + sRow) * lda + sChunk;
    const half_t* bSrc = Bt + (size_t)(bn0 + sRow) * K   + sChunk;

    for (int k0 = 0; k0 < K; k0 += BK) {
        v16h av = *(const v16h*)(aSrc + k0);
        v16h bv = *(const v16h*)(bSrc + k0);
        if (k0 + BK < K) {   // global_prefetch_b8 of next tiles
            __builtin_prefetch(aSrc + k0 + BK, 0, 1);
            __builtin_prefetch(bSrc + k0 + BK, 0, 1);
        }

        __syncthreads();  // protect previous iteration's fragment reads
        *(v16h*)&As[sRow * LDA_S + sChunk] = av;
        *(v16h*)&Bs[sRow * LDB_S + sChunk] = bv;
        __syncthreads();

        // A fragments: lane<16 -> K0..15 of row m; lane>=16 -> K16..31
        v16h af[4];
#pragma unroll
        for (int mt = 0; mt < 4; ++mt) {
            const int m  = wm * 64 + mt * 16 + lrow;
            const int kb = lhalf * 16;
            v8h alo = *(const v8h*)&As[m * LDA_S + kb];
            v8h ahi = *(const v8h*)&As[m * LDA_S + kb + 8];
            af[mt] = __builtin_shufflevector(alo, ahi,
                                             0, 1, 2, 3, 4, 5, 6, 7,
                                             8, 9, 10, 11, 12, 13, 14, 15);
        }
        // B fragments: lane<16 -> K0..15 of column n; lane>=16 -> K16..31
        v16h bf[2];
#pragma unroll
        for (int nt = 0; nt < 2; ++nt) {
            const int nn = wn * 32 + nt * 16 + lrow;
            bf[nt] = *(const v16h*)&Bs[nn * LDB_S + lhalf * 16];
        }

#pragma unroll
        for (int mt = 0; mt < 4; ++mt)
#pragma unroll
            for (int nt = 0; nt < 2; ++nt)
                acc[mt][nt] = __builtin_amdgcn_wmma_f32_16x16x32_f16(
                    false, af[mt], false, bf[nt], (short)0, acc[mt][nt], false, false);
    }

    // epilogue: C tile layout -> VGPR r: lane<16 (M=r,N=lane), lane>=16 (M=8+r,N=lane-16)
#pragma unroll
    for (int mt = 0; mt < 4; ++mt) {
#pragma unroll
        for (int nt = 0; nt < 2; ++nt) {
            const int gn  = bn0 + wn * 32 + nt * 16 + lrow;
            const int gm0 = bm0 + wm * 64 + mt * 16 + lhalf * 8;
            const float bval = bias[gn];
#pragma unroll
            for (int r = 0; r < 8; ++r) {
                const int gm = gm0 + r;
                float v = acc[mt][nt][r] + bval;
                if (MODE == 0) {
                    outH[(size_t)gm * Ncols + gn] = (half_t)v;
                } else if (MODE == 1) {
                    outH[(size_t)gm * Ncols + gn] = (half_t)gelu_tanh(v);
                } else {
                    outF[(size_t)gm * Ncols + gn] = v + resid[(size_t)gm * Ncols + gn];
                }
            }
        }
    }
}

// ---------------------------------------------------------------------------
// Sparse attention + relative-position features.
// One block (128 threads) per query node; thread t = (head h, edge slot j).
// Edges for node n are contiguous: e = n*DEG + j  (row_index = repeat(arange)).
// Writes f16 feat into h_buf[:, 512:536].
// ---------------------------------------------------------------------------
__global__ __launch_bounds__(128)
void attn_kernel(const half_t* __restrict__ q, const half_t* __restrict__ k,
                 const float* __restrict__ pos, const float* __restrict__ src_pos,
                 const float* __restrict__ att_bias,
                 const int* __restrict__ src_index, const int* __restrict__ org_to_src,
                 half_t* __restrict__ h_buf) {
    const int n = blockIdx.x;
    const int t = threadIdx.x;
    const int h = t >> 4;
    const int j = t & 15;

    __shared__ half_t qs[EMB];
    __shared__ float ls[128];
    __shared__ float cxs[128], cys[128], czs[128];

    // cooperative load of this node's q row (1 KB)
    if (t < 64) ((uint4*)qs)[t] = ((const uint4*)(q + (size_t)n * EMB))[t];
    __syncthreads();

    const int e    = n * DEG + j;
    const int s    = src_index[e];
    const int node = org_to_src[s];

    const half_t* qr = qs + h * HDIM;
    const half_t* kr = k + (size_t)node * EMB + h * HDIM;
    float dot = 0.f;
#pragma unroll
    for (int d = 0; d < HDIM; ++d) dot += (float)qr[d] * (float)kr[d];
    const float logit = dot * SCALE + att_bias[e];

    ls[t] = logit; __syncthreads();
    float mx = -INFINITY;
#pragma unroll
    for (int i = 0; i < DEG; ++i) mx = fmaxf(mx, ls[h * DEG + i]);
    const float w = __expf(logit - mx);
    __syncthreads(); ls[t] = w; __syncthreads();
    float sum = 0.f;
#pragma unroll
    for (int i = 0; i < DEG; ++i) sum += ls[h * DEG + i];
    const float attn = w / sum;

    cxs[t] = attn * src_pos[(size_t)s * 3 + 0];
    cys[t] = attn * src_pos[(size_t)s * 3 + 1];
    czs[t] = attn * src_pos[(size_t)s * 3 + 2];
    __syncthreads();

    if (j < 3) {
        const float* arr = (j == 0) ? cxs : (j == 1) ? cys : czs;
        float a = 0.f;
#pragma unroll
        for (int i = 0; i < DEG; ++i) a += arr[h * DEG + i];
        a -= pos[(size_t)n * 3 + j];
        h_buf[(size_t)n * KPAD + EMB + h * 3 + j] = (half_t)a;
    }
}

// ---------------------------------------------------------------------------
// launcher
// ---------------------------------------------------------------------------
static inline size_t align256(size_t v) { return (v + 255) & ~(size_t)255; }

extern "C" void kernel_launch(void* const* d_in, const int* in_sizes, int n_in,
                              void* d_out, int out_size, void* d_ws, size_t ws_size,
                              hipStream_t stream) {
    const float* x        = (const float*)d_in[0];
    const float* pos      = (const float*)d_in[1];
    const float* src_pos  = (const float*)d_in[2];
    const float* att_bias = (const float*)d_in[3];
    const float* Wq       = (const float*)d_in[4];
    const float* bq       = (const float*)d_in[5];
    const float* Wk       = (const float*)d_in[6];
    const float* bk       = (const float*)d_in[7];
    const float* g_att    = (const float*)d_in[8];
    const float* b_att    = (const float*)d_in[9];
    const float* g_mlp    = (const float*)d_in[10];
    const float* b_mlp    = (const float*)d_in[11];
    const float* W1       = (const float*)d_in[12];
    const float* b1       = (const float*)d_in[13];
    const float* W2       = (const float*)d_in[14];
    const float* b2       = (const float*)d_in[15];
    const int*   src_index  = (const int*)d_in[17];
    const int*   org_to_src = (const int*)d_in[18];
    float* out = (float*)d_out;

    // carve workspace (~106 MB of f16 buffers)
    char* w = (char*)d_ws;
    auto carve = [&](size_t halves) {
        half_t* p = (half_t*)w;
        w += align256(halves * sizeof(half_t));
        return p;
    };
    half_t* z_att = carve((size_t)N_NODES * EMB);
    half_t* q_h   = carve((size_t)N_NODES * EMB);
    half_t* k_h   = carve((size_t)N_NODES * EMB);
    half_t* h_buf = carve((size_t)N_NODES * KPAD);
    half_t* act   = carve((size_t)N_NODES * HID);
    half_t* WqT   = carve((size_t)EMB * EMB);    // [512, 512]
    half_t* WkT   = carve((size_t)EMB * EMB);    // [512, 512]
    half_t* W1T   = carve((size_t)HID * KPAD);   // [1024, 544]
    half_t* W2T   = carve((size_t)EMB * HID);    // [512, 1024]

    // 1) weight conversion + transpose (independent, tiny)
    {
        int tot = EMB * EMB;
        cvt_wT_kernel<<<(tot + 255) / 256, 256, 0, stream>>>(Wq, WqT, EMB, EMB, EMB);
        cvt_wT_kernel<<<(tot + 255) / 256, 256, 0, stream>>>(Wk, WkT, EMB, EMB, EMB);
        int tot1 = HID * KPAD;
        cvt_wT_kernel<<<(tot1 + 255) / 256, 256, 0, stream>>>(W1, W1T, EMB + 3 * NHEAD, KPAD, HID);
        int tot2 = EMB * HID;
        cvt_wT_kernel<<<(tot2 + 255) / 256, 256, 0, stream>>>(W2, W2T, HID, HID, EMB);
    }

    // 2) dual layernorm
    ln_dual_kernel<<<N_NODES, 128, 0, stream>>>(x, g_att, b_att, g_mlp, b_mlp, z_att, h_buf);

    // 3) Q / K projections (WMMA)
    {
        dim3 grid(N_NODES / BM, EMB / BN);
        gemm_f16_kernel<0><<<grid, 256, 0, stream>>>(z_att, EMB, WqT, bq, nullptr,
                                                     q_h, nullptr, N_NODES, EMB, EMB);
        gemm_f16_kernel<0><<<grid, 256, 0, stream>>>(z_att, EMB, WkT, bk, nullptr,
                                                     k_h, nullptr, N_NODES, EMB, EMB);
    }

    // 4) sparse attention -> feat columns of h_buf
    attn_kernel<<<N_NODES, 128, 0, stream>>>(q_h, k_h, pos, src_pos, att_bias,
                                             src_index, org_to_src, h_buf);

    // 5) MLP layer 1: gelu(h @ W1 + b1) (WMMA)
    {
        dim3 grid(N_NODES / BM, HID / BN);
        gemm_f16_kernel<1><<<grid, 256, 0, stream>>>(h_buf, KPAD, W1T, b1, nullptr,
                                                     act, nullptr, N_NODES, HID, KPAD);
    }

    // 6) MLP layer 2 + residual + b2 -> fp32 out (WMMA)
    {
        dim3 grid(N_NODES / BM, EMB / BN);
        gemm_f16_kernel<2><<<grid, 256, 0, stream>>>(act, HID, W2T, b2, x,
                                                     nullptr, out, N_NODES, EMB, HID);
    }
    (void)in_sizes; (void)n_in; (void)out_size; (void)ws_size;
}
